// MMCL_HardNegative_47330539601986
// MI455X (gfx1250) — compile-verified, matrix operand order checked by reading, not run on hardware
//
#include <hip/hip_runtime.h>
#include <hip/hip_bf16.h>
#include <math.h>

// ---------------- problem constants ----------------
#define BS     8192
#define DIM    256
#define KNEG   16
#define GAMMA  (1.0f / 0.07f)

// ---------------- tiling constants -----------------
#define ROWS_PER_BLOCK 128              // 8 waves x 16 rows
#define NCHUNK         32               // columns per LDS chunk (double-buffered)
#define NCHUNKS        (BS / NCHUNK)    // 256
#define LDSK           (DIM + 8)        // padded bf16 row stride (264)
#define BUFELEMS       (NCHUNK * LDSK)  // bf16 elements per buffer (8448)
#define TSTR           20               // tile row stride (floats): 16B-aligned, conflict-free
#define NBLOCKS        (BS / ROWS_PER_BLOCK)   // 64

typedef __attribute__((ext_vector_type(16))) __bf16 v16bf;
typedef __attribute__((ext_vector_type(8)))  __bf16 v8bf;
typedef __attribute__((ext_vector_type(8)))  float  v8f;

__device__ __forceinline__ v8bf pack8(float4 a, float4 b) {
    v8bf p;
    p[0] = (__bf16)a.x; p[1] = (__bf16)a.y; p[2] = (__bf16)a.z; p[3] = (__bf16)a.w;
    p[4] = (__bf16)b.x; p[5] = (__bf16)b.y; p[6] = (__bf16)b.z; p[7] = (__bf16)b.w;
    return p;
}

__device__ __forceinline__ v16bf cat16(v8bf lo, v8bf hi) {
    return __builtin_shufflevector(lo, hi, 0,1,2,3,4,5,6,7,8,9,10,11,12,13,14,15);
}

__device__ __forceinline__ v16bf ldb(const __bf16* p) {
    v8bf lo = *(const v8bf*)(p);
    v8bf hi = *(const v8bf*)(p + 16);
    return cat16(lo, hi);
}

__device__ __forceinline__ float max4(float4 v) {
    return fmaxf(fmaxf(v.x, v.y), fmaxf(v.z, v.w));
}

// sorted-ascending top-16 insert: replace min then compare-swap chain
#define TRY_INSERT(val)                                              \
    do {                                                             \
        float _v = (val);                                            \
        if (_v > best[0]) {                                          \
            best[0] = _v;                                            \
            _Pragma("unroll")                                        \
            for (int _q = 0; _q < KNEG - 1; ++_q) {                  \
                float _x = best[_q], _y = best[_q + 1];              \
                best[_q]     = fminf(_x, _y);                        \
                best[_q + 1] = fmaxf(_x, _y);                        \
            }                                                        \
        }                                                            \
    } while (0)

// Fused: S = F1 @ F2^T via dual-stream WMMA bf16 with double-buffered LDS
// staging; per-row top-16 of raw dot products (exp is monotone) with tile-max
// fast path; exact fp32 diagonal; deterministic two-stage reduction.
__global__ void __launch_bounds__(256)
mmcl_hardneg_main(const float* __restrict__ feat, float* __restrict__ ws) {
    __shared__ __align__(16) __bf16 Bs[2 * BUFELEMS];     // 33,792 B double-buffered F2 (bf16)
    __shared__ __align__(16) float Tile[16 * 16 * TSTR];  // 20,480 B: 2 tiles x 8 waves
    __shared__ float RedP[256];
    __shared__ float RedN[256];

    const int t    = threadIdx.x;
    const int wv   = t >> 5;          // wave id 0..7 (wave32)
    const int lane = t & 31;
    const int lrow = lane & 15;       // row-within-wave / col-within-subtile
    const int koff = (lane >> 4) * 8; // WMMA 16-bit operand K sub-offset per half-wave

    const int rowBase = blockIdx.x * ROWS_PER_BLOCK + wv * 16;   // wave-uniform
    const int row     = rowBase + lrow;                          // owned by lanes 0-15

    // ---- A fragments: this wave's 16 rows, all of K=256, WMMA bf16 layout ----
    v16bf af[8];
    {
        const float* f1 = feat + (size_t)row * (2 * DIM) + koff;
        #pragma unroll
        for (int ks = 0; ks < 8; ++ks) {
            const float* p = f1 + ks * 32;
            float4 a0 = *(const float4*)(p +  0);
            float4 a1 = *(const float4*)(p +  4);
            float4 a2 = *(const float4*)(p + 16);
            float4 a3 = *(const float4*)(p + 20);
            af[ks] = cat16(pack8(a0, a1), pack8(a2, a3));
        }
    }

    // ---- exact fp32 positive pair: diag = dot(F1[row], F2[row]) (once) ----
    float diag = 0.0f;
    if (lane < 16) {
        const float* p1 = feat + (size_t)row * (2 * DIM);
        const float* p2 = p1 + DIM;
        float acc = 0.0f;
        #pragma unroll 4
        for (int i = 0; i < DIM; i += 4) {
            float4 a = *(const float4*)(p1 + i);
            float4 b = *(const float4*)(p2 + i);
            acc = fmaf(a.x, b.x, acc); acc = fmaf(a.y, b.y, acc);
            acc = fmaf(a.z, b.z, acc); acc = fmaf(a.w, b.w, acc);
        }
        diag = acc;
    }

    // per-row top-16 of raw dot products, ascending (best[0] = running min)
    float best[KNEG];
    #pragma unroll
    for (int q = 0; q < KNEG; ++q) best[q] = -3.0e38f;

    // merge helper: 16 candidates of owned row from a spilled tile
    auto merge_tile = [&](const float* rp, int jg0) {
        float4 r0 = *(const float4*)(rp + 0);
        float4 r1 = *(const float4*)(rp + 4);
        float4 r2 = *(const float4*)(rp + 8);
        float4 r3 = *(const float4*)(rp + 12);
        if (jg0 != rowBase) {
            // common path: tile-max gate, then rare insert chain
            float tmax = fmaxf(fmaxf(max4(r0), max4(r1)), fmaxf(max4(r2), max4(r3)));
            if (tmax > best[0]) {
                TRY_INSERT(r0.x); TRY_INSERT(r0.y); TRY_INSERT(r0.z); TRY_INSERT(r0.w);
                TRY_INSERT(r1.x); TRY_INSERT(r1.y); TRY_INSERT(r1.z); TRY_INSERT(r1.w);
                TRY_INSERT(r2.x); TRY_INSERT(r2.y); TRY_INSERT(r2.z); TRY_INSERT(r2.w);
                TRY_INSERT(r3.x); TRY_INSERT(r3.y); TRY_INSERT(r3.z); TRY_INSERT(r3.w);
            }
        } else {
            // diagonal tile: taken exactly once per wave over the whole kernel
            float cand[16] = { r0.x, r0.y, r0.z, r0.w, r1.x, r1.y, r1.z, r1.w,
                               r2.x, r2.y, r2.z, r2.w, r3.x, r3.y, r3.z, r3.w };
            #pragma unroll
            for (int j = 0; j < 16; ++j)
                if (j == lrow) cand[j] = -3.0e38f;   // mask positive pair
            #pragma unroll
            for (int j = 0; j < 16; ++j) TRY_INSERT(cand[j]);
        }
    };

    // staging assignment: 8 threads per column, each covers 32 consecutive K
    const int scol = t >> 3;          // 0..31
    const int sk   = (t & 7) * 32;    // 0,32,...,224

    // prologue: stage chunk 0 into buffer 0
    {
        const float* f2 = feat + (size_t)scol * (2 * DIM) + DIM + sk;
        float4 s0 = *(const float4*)(f2 +  0), s1 = *(const float4*)(f2 +  4);
        float4 s2 = *(const float4*)(f2 +  8), s3 = *(const float4*)(f2 + 12);
        float4 s4 = *(const float4*)(f2 + 16), s5 = *(const float4*)(f2 + 20);
        float4 s6 = *(const float4*)(f2 + 24), s7 = *(const float4*)(f2 + 28);
        __bf16* l = Bs + scol * LDSK + sk;
        *(v8bf*)(l +  0) = pack8(s0, s1);
        *(v8bf*)(l +  8) = pack8(s2, s3);
        *(v8bf*)(l + 16) = pack8(s4, s5);
        *(v8bf*)(l + 24) = pack8(s6, s7);
    }
    __syncthreads();

    for (int n = 0; n < NCHUNKS; ++n) {
        const int c0   = n * NCHUNK;
        const int cur  = n & 1;
        const bool pf  = (n + 1 < NCHUNKS);
        const float* f2n = feat + (size_t)(c0 + NCHUNK + scol) * (2 * DIM) + DIM + sk;

        // prefetch batch A of next chunk (16 VGPRs), flies under the WMMA loop
        float4 sA0, sA1, sA2, sA3;
        if (pf) {
            sA0 = *(const float4*)(f2n +  0); sA1 = *(const float4*)(f2n +  4);
            sA2 = *(const float4*)(f2n +  8); sA3 = *(const float4*)(f2n + 12);
        }

        // ---- dual-stream WMMA: both 16-col sub-tiles interleaved ----
        const __bf16* bp0 = Bs + cur * BUFELEMS + lrow * LDSK + koff;
        const __bf16* bp1 = bp0 + 16 * LDSK;

        v16bf b0 = ldb(bp0);
        v16bf b1 = ldb(bp1);
        v8f c0v = {};
        v8f c1v = {};
        #pragma unroll
        for (int ks = 0; ks < 8; ++ks) {
            v16bf n0, n1;
            if (ks < 7) {
                n0 = ldb(bp0 + (ks + 1) * 32);
                n1 = ldb(bp1 + (ks + 1) * 32);
            }
            c0v = __builtin_amdgcn_wmma_f32_16x16x32_bf16(
                      false, af[ks], false, b0, (short)0, c0v, false, false);
            c1v = __builtin_amdgcn_wmma_f32_16x16x32_bf16(
                      false, af[ks], false, b1, (short)0, c1v, false, false);
            if (ks < 7) { b0 = n0; b1 = n1; }
        }

        // prefetch batch B of next chunk
        float4 sB0, sB1, sB2, sB3;
        if (pf) {
            sB0 = *(const float4*)(f2n + 16); sB1 = *(const float4*)(f2n + 20);
            sB2 = *(const float4*)(f2n + 24); sB3 = *(const float4*)(f2n + 28);
        }

        // ---- spill both tiles: lane<16 -> rows 0..7, lane>=16 -> rows 8..15 ----
        float* tp0 = Tile + wv * (16 * TSTR);
        float* tp1 = tp0 + 8 * (16 * TSTR);
        const int mbase = (lane >> 4) * 8;
        #pragma unroll
        for (int v = 0; v < 8; ++v) {
            tp0[(mbase + v) * TSTR + lrow] = c0v[v];
            tp1[(mbase + v) * TSTR + lrow] = c1v[v];
        }
        __builtin_amdgcn_wave_barrier();   // keep ds store->load program order

        // ---- merge both tiles (lanes 0..15 only) ----
        if (lane < 16) {
            merge_tile(tp0 + lrow * TSTR, c0);
            merge_tile(tp1 + lrow * TSTR, c0 + 16);
        }
        __builtin_amdgcn_wave_barrier();

        // convert + store next chunk into the other buffer (loads have landed)
        if (pf) {
            __bf16* l = Bs + (cur ^ 1) * BUFELEMS + scol * LDSK + sk;
            *(v8bf*)(l +  0) = pack8(sA0, sA1);
            *(v8bf*)(l +  8) = pack8(sA2, sA3);
            *(v8bf*)(l + 16) = pack8(sB0, sB1);
            *(v8bf*)(l + 24) = pack8(sB2, sB3);
        }
        __syncthreads();
    }

    // ---- apply exp to survivors, block reduction ----
    float psum = 0.0f, nsum = 0.0f;
    if (lane < 16) {
        psum = __expf(GAMMA * (2.0f * diag - 2.0f));
        #pragma unroll
        for (int q = 0; q < KNEG; ++q)
            nsum += __expf(GAMMA * (2.0f * best[q] - 2.0f));
    }
    RedP[t] = psum;
    RedN[t] = nsum;
    __syncthreads();
    if (t == 0) {
        float p = 0.0f, n = 0.0f;
        for (int i = 0; i < 256; ++i) { p += RedP[i]; n += RedN[i]; }
        ws[2 * blockIdx.x]     = p;
        ws[2 * blockIdx.x + 1] = n;
    }
}

// Deterministic final fold of 64 block partials -> two output scalars.
__global__ void mmcl_hardneg_finalize(const float* __restrict__ ws, float* __restrict__ out) {
    if (threadIdx.x == 0) {
        float p = 0.0f, n = 0.0f;
        for (int i = 0; i < NBLOCKS; ++i) { p += ws[2 * i]; n += ws[2 * i + 1]; }
        out[0] = -(p / (float)BS);                     // -pos_loss
        out[1] =  n / ((float)BS * (float)KNEG);       // hard_neg_loss
    }
}

extern "C" void kernel_launch(void* const* d_in, const int* in_sizes, int n_in,
                              void* d_out, int out_size, void* d_ws, size_t ws_size,
                              hipStream_t stream) {
    (void)in_sizes; (void)n_in; (void)out_size; (void)ws_size;
    const float* feat = (const float*)d_in[0];   // (8192, 2, 256) f32
    float* out = (float*)d_out;                  // 2 floats
    float* ws  = (float*)d_ws;                   // >= 64*2 floats

    mmcl_hardneg_main<<<NBLOCKS, 256, 0, stream>>>(feat, ws);
    mmcl_hardneg_finalize<<<1, 32, 0, stream>>>(ws, out);
}
